// Attention_1683627180827
// MI455X (gfx1250) — matrix operand order repaired and checked by reading, not compiled, not run on hardware
//
#include <hip/hip_runtime.h>

// ---------- CDNA5 (gfx1250) wave32 WMMA attention ----------
// Matrix ops: v_wmma_f32_16x16x32_f16 (D = A16x32 * B32x16 + C, f32 acc)
// Tile fills: global_load_async_to_lds_b128 (ASYNCcnt) on pre-converted f16

typedef __attribute__((ext_vector_type(16))) _Float16 v16h;
typedef __attribute__((ext_vector_type(8)))  _Float16 v8h;
typedef __attribute__((ext_vector_type(4)))  _Float16 v4h;
typedef __attribute__((ext_vector_type(8)))  float    v8f;

union F16x16 { v16h v; v8h h[2]; v4h q[4]; };

#define CIN   768
#define C3    2304
#define NHEAD 12
#define NSEQ  1024
#define HDIM  64
#define ASTR  40   // LDS row stride (f16): 80B rows -> 16B aligned, 16 rows hit 16 banks
#define BSTR  40
#define PSTR  132  // P staging stride (f16)

static __device__ __forceinline__ uint32_t lds_off(const void* p) {
    return (uint32_t)(uintptr_t)p;   // low 32 bits of shared-aperture flat addr = LDS offset
}

static __device__ __forceinline__ void async_b128(uint32_t lds, const void* gp) {
    asm volatile("global_load_async_to_lds_b128 %0, %1, off"
                 :: "v"(lds), "v"(gp) : "memory");
}

static __device__ __forceinline__ void wait_async0() {
    asm volatile("s_wait_asynccnt 0" ::: "memory");
}

// ---------------------------------------------------------------------------
// K0a: f32 -> f16 straight convert (x -> xh), 4 elems/thread
// ---------------------------------------------------------------------------
__global__ __launch_bounds__(256)
void cvt_copy(const float* __restrict__ src, _Float16* __restrict__ dst)
{
    int i = (blockIdx.x * 256 + threadIdx.x) * 4;
    float4 v = *(const float4*)(src + i);
    v4h o; o.x = (_Float16)v.x; o.y = (_Float16)v.y; o.z = (_Float16)v.z; o.w = (_Float16)v.w;
    *(v4h*)(dst + i) = o;
}

// ---------------------------------------------------------------------------
// K0b: f32 -> f16 transpose: src[rows][cols] -> dst[cols][rows]
// ---------------------------------------------------------------------------
__global__ __launch_bounds__(256)
void cvt_transpose(const float* __restrict__ src, _Float16* __restrict__ dst,
                   int rows, int cols)
{
    int idx = blockIdx.x * 256 + threadIdx.x;     // coalesced on dst
    int c = idx / rows, r = idx - c * rows;
    dst[(size_t)c * rows + r] = (_Float16)src[(size_t)r * cols + c];
}

// ---------------------------------------------------------------------------
// K1: qkv = xh @ wqkvT^T  -> scatter f16 to Q[B,H,N,D], K[B,H,N,D], V^T[B,H,D,N]
// 128x128 block tile, 8 waves (4x2), each 32x64; K-step 32; async LDS fills.
// ---------------------------------------------------------------------------
__global__ __launch_bounds__(256)
void qkv_gemm(const _Float16* __restrict__ xh, const _Float16* __restrict__ wT,
              _Float16* __restrict__ Qb, _Float16* __restrict__ Kb,
              _Float16* __restrict__ Vt)
{
    __shared__ alignas(16) _Float16 Al[128 * ASTR];
    __shared__ alignas(16) _Float16 Bl[128 * BSTR];

    const int tid  = threadIdx.x;
    const int wave = tid >> 5;
    const int lane = tid & 31;
    const int l16  = lane & 15;
    const int hi16 = lane >> 4;
    const int m0   = blockIdx.x * 128;
    const int c0   = blockIdx.y * 128;
    const int wr   = (wave & 3) * 32;
    const int wc   = (wave >> 2) * 64;

    // async copy decomposition: 128 rows x 32 f16 = 512 x 16B chunks, 2/thread
    const int r0 = tid >> 2;               // 0..63
    const int c8 = (tid & 3) * 8;          // 0,8,16,24
    const uint32_t alb = lds_off(Al), blb = lds_off(Bl);

    v8f acc[2][4] = {};

    for (int k0 = 0; k0 < CIN; k0 += 32) {
        __syncthreads();
        #pragma unroll
        for (int p = 0; p < 2; ++p) {
            int r = r0 + p * 64;
            async_b128(alb + (uint32_t)(r * ASTR + c8) * 2,
                       xh + (size_t)(m0 + r) * CIN + k0 + c8);
            async_b128(blb + (uint32_t)(r * BSTR + c8) * 2,
                       wT + (size_t)(c0 + r) * CIN + k0 + c8);
        }
        wait_async0();
        __syncthreads();

        F16x16 bf[4];
        #pragma unroll
        for (int ct = 0; ct < 4; ++ct) {
            const _Float16* bp = &Bl[(wc + ct * 16 + l16) * BSTR + hi16 * 8];
            bf[ct].h[0] = *(const v8h*)(bp);
            bf[ct].h[1] = *(const v8h*)(bp + 16);
        }
        #pragma unroll
        for (int rt = 0; rt < 2; ++rt) {
            const _Float16* ap = &Al[(wr + rt * 16 + l16) * ASTR + hi16 * 8];
            F16x16 af;
            af.h[0] = *(const v8h*)(ap);
            af.h[1] = *(const v8h*)(ap + 16);
            #pragma unroll
            for (int ct = 0; ct < 4; ++ct) {
                acc[rt][ct] = __builtin_amdgcn_wmma_f32_16x16x32_f16(
                    false, af.v, false, bf[ct].v, (short)0, acc[rt][ct], false, false);
            }
        }
    }

    #pragma unroll
    for (int rt = 0; rt < 2; ++rt) {
        #pragma unroll
        for (int ct = 0; ct < 4; ++ct) {
            #pragma unroll
            for (int v = 0; v < 8; ++v) {
                int m = m0 + wr + rt * 16 + v + hi16 * 8;
                int e = c0 + wc + ct * 16 + l16;
                _Float16 hv = (_Float16)acc[rt][ct][v];
                int b = m >> 10, n = m & 1023;
                int three = e / CIN;
                int rem = e - three * CIN;
                int hh = rem >> 6, d = rem & 63;
                size_t bh = (size_t)(b * NHEAD + hh);
                if (three == 0)      Qb[(bh * NSEQ + n) * HDIM + d] = hv;
                else if (three == 1) Kb[(bh * NSEQ + n) * HDIM + d] = hv;
                else                 Vt[(bh * HDIM + d) * NSEQ + n] = hv;
            }
        }
    }
}

// ---------------------------------------------------------------------------
// K2: flash attention. Block = 128 query rows of one (b,h); wave = 16 rows.
// K/V fragments load straight from global in WMMA layout; P transposed via LDS.
// ---------------------------------------------------------------------------
__global__ __launch_bounds__(256)
void attn_fwd(const _Float16* __restrict__ Qb, const _Float16* __restrict__ Kb,
              const _Float16* __restrict__ Vt, _Float16* __restrict__ Ab)
{
    __shared__ alignas(16) _Float16 Pl[8][16 * PSTR];

    const int tid  = threadIdx.x;
    const int wave = tid >> 5;
    const int lane = tid & 31;
    const int l16  = lane & 15;
    const int hi16 = lane >> 4;
    const int bh   = blockIdx.x;
    const int b    = bh / NHEAD;
    const int h    = bh - b * NHEAD;
    const int m0   = blockIdx.y * 128 + wave * 16;
    const size_t base = (size_t)bh * NSEQ * HDIM;

    F16x16 qf[2];
    #pragma unroll
    for (int ch = 0; ch < 2; ++ch) {
        const _Float16* qp = Qb + base + (size_t)(m0 + l16) * HDIM + ch * 32 + hi16 * 16;
        qf[ch].h[0] = *(const v8h*)qp;
        qf[ch].h[1] = *(const v8h*)(qp + 8);
    }

    float mrow[8], lrow[8];
    #pragma unroll
    for (int v = 0; v < 8; ++v) { mrow[v] = -3.0e38f; lrow[v] = 0.0f; }
    v8f oacc[4] = {};

    _Float16* myP = &Pl[wave][0];
    const float scale = 0.125f;   // 1/sqrt(64)

    for (int nb = 0; nb < NSEQ; nb += 128) {
        v8f s[8];
        #pragma unroll
        for (int t = 0; t < 8; ++t) {
            v8f a = {};
            #pragma unroll
            for (int ch = 0; ch < 2; ++ch) {
                const _Float16* kp = Kb + base + (size_t)(nb + t * 16 + l16) * HDIM
                                        + ch * 32 + hi16 * 16;
                v16h bfr = *(const v16h*)kp;
                a = __builtin_amdgcn_wmma_f32_16x16x32_f16(
                        false, qf[ch].v, false, bfr, (short)0, a, false, false);
            }
            s[t] = a;
        }

        #pragma unroll
        for (int v = 0; v < 8; ++v) {
            float mx = -3.0e38f;
            #pragma unroll
            for (int t = 0; t < 8; ++t) {
                float sv = s[t][v] * scale;
                s[t][v] = sv;
                mx = fmaxf(mx, sv);
            }
            #pragma unroll
            for (int off = 1; off < 16; off <<= 1)
                mx = fmaxf(mx, __shfl_xor(mx, off, 32));
            float mnew = fmaxf(mrow[v], mx);
            float corr = __expf(mrow[v] - mnew);
            mrow[v] = mnew;
            lrow[v] *= corr;
            #pragma unroll
            for (int t = 0; t < 4; ++t) oacc[t][v] *= corr;
            float ps = 0.0f;
            #pragma unroll
            for (int t = 0; t < 8; ++t) {
                float p = __expf(s[t][v] - mnew);
                ps += p;
                myP[(v + hi16 * 8) * PSTR + t * 16 + l16] = (_Float16)p;
            }
            #pragma unroll
            for (int off = 1; off < 16; off <<= 1)
                ps += __shfl_xor(ps, off, 32);
            lrow[v] += ps;
        }

        // wave-local LDS transpose: wait for this wave's ds_stores before re-reading
        asm volatile("s_wait_dscnt 0" ::: "memory");

        F16x16 pf[4];
        #pragma unroll
        for (int kc = 0; kc < 4; ++kc) {
            const _Float16* pp = &myP[l16 * PSTR + kc * 32 + hi16 * 16];
            pf[kc].q[0] = *(const v4h*)(pp + 0);
            pf[kc].q[1] = *(const v4h*)(pp + 4);
            pf[kc].q[2] = *(const v4h*)(pp + 8);
            pf[kc].q[3] = *(const v4h*)(pp + 12);
        }

        #pragma unroll
        for (int t2 = 0; t2 < 4; ++t2) {
            v8f o = oacc[t2];
            #pragma unroll
            for (int kc = 0; kc < 4; ++kc) {
                const _Float16* vp = Vt + base + (size_t)(t2 * 16 + l16) * NSEQ
                                        + nb + kc * 32 + hi16 * 16;
                v16h vfr = *(const v16h*)vp;
                o = __builtin_amdgcn_wmma_f32_16x16x32_f16(
                        false, pf[kc].v, false, vfr, (short)0, o, false, false);
            }
            oacc[t2] = o;
        }
    }

    #pragma unroll
    for (int t2 = 0; t2 < 4; ++t2) {
        #pragma unroll
        for (int v = 0; v < 8; ++v) {
            int n = m0 + v + hi16 * 8;
            int d = t2 * 16 + l16;
            float o = oacc[t2][v] / lrow[v];
            Ab[(size_t)(b * NSEQ + n) * CIN + h * HDIM + d] = (_Float16)o;
        }
    }
}

// ---------------------------------------------------------------------------
// K3: out = Ab(f16) @ wprojT^T + bias   (f32 output), async LDS fills
// ---------------------------------------------------------------------------
__global__ __launch_bounds__(256)
void proj_gemm(const _Float16* __restrict__ Ab, const _Float16* __restrict__ wT,
               const float* __restrict__ bias, float* __restrict__ out)
{
    __shared__ alignas(16) _Float16 Al[128 * ASTR];
    __shared__ alignas(16) _Float16 Bl[128 * BSTR];

    const int tid  = threadIdx.x;
    const int wave = tid >> 5;
    const int lane = tid & 31;
    const int l16  = lane & 15;
    const int hi16 = lane >> 4;
    const int m0   = blockIdx.x * 128;
    const int c0   = blockIdx.y * 128;
    const int wr   = (wave & 3) * 32;
    const int wc   = (wave >> 2) * 64;

    const int r0 = tid >> 2;
    const int c8 = (tid & 3) * 8;
    const uint32_t alb = lds_off(Al), blb = lds_off(Bl);

    v8f acc[2][4] = {};

    for (int k0 = 0; k0 < CIN; k0 += 32) {
        __syncthreads();
        #pragma unroll
        for (int p = 0; p < 2; ++p) {
            int r = r0 + p * 64;
            async_b128(alb + (uint32_t)(r * ASTR + c8) * 2,
                       Ab + (size_t)(m0 + r) * CIN + k0 + c8);
            async_b128(blb + (uint32_t)(r * BSTR + c8) * 2,
                       wT + (size_t)(c0 + r) * CIN + k0 + c8);
        }
        wait_async0();
        __syncthreads();

        F16x16 bf[4];
        #pragma unroll
        for (int ct = 0; ct < 4; ++ct) {
            const _Float16* bp = &Bl[(wc + ct * 16 + l16) * BSTR + hi16 * 8];
            bf[ct].h[0] = *(const v8h*)(bp);
            bf[ct].h[1] = *(const v8h*)(bp + 16);
        }
        #pragma unroll
        for (int rt = 0; rt < 2; ++rt) {
            const _Float16* ap = &Al[(wr + rt * 16 + l16) * ASTR + hi16 * 8];
            F16x16 af;
            af.h[0] = *(const v8h*)(ap);
            af.h[1] = *(const v8h*)(ap + 16);
            #pragma unroll
            for (int ct = 0; ct < 4; ++ct) {
                acc[rt][ct] = __builtin_amdgcn_wmma_f32_16x16x32_f16(
                    false, af.v, false, bf[ct].v, (short)0, acc[rt][ct], false, false);
            }
        }
    }

    #pragma unroll
    for (int rt = 0; rt < 2; ++rt) {
        #pragma unroll
        for (int ct = 0; ct < 4; ++ct) {
            int e = c0 + wc + ct * 16 + l16;
            float bv = bias[e];
            #pragma unroll
            for (int v = 0; v < 8; ++v) {
                int m = m0 + wr + rt * 16 + v + hi16 * 8;
                out[(size_t)m * CIN + e] = acc[rt][ct][v] + bv;
            }
        }
    }
}

// ---------------------------------------------------------------------------
extern "C" void kernel_launch(void* const* d_in, const int* in_sizes, int n_in,
                              void* d_out, int out_size, void* d_ws, size_t ws_size,
                              hipStream_t stream)
{
    (void)in_sizes; (void)n_in; (void)out_size; (void)ws_size;
    const float* x     = (const float*)d_in[0];
    const float* wqkv  = (const float*)d_in[1];
    const float* wproj = (const float*)d_in[2];
    const float* bproj = (const float*)d_in[3];
    float* out = (float*)d_out;

    const size_t SZ = (size_t)8 * NHEAD * NSEQ * HDIM;  // 6,291,456
    _Float16* Qb  = (_Float16*)d_ws;
    _Float16* Kb  = Qb + SZ;
    _Float16* Vt  = Kb + SZ;
    _Float16* xh  = Vt + SZ;          // x as f16; later reused as Ab (K1 is done with it)
    _Float16* Ab  = xh;
    _Float16* wqT = xh + SZ;          // [2304][768]
    _Float16* wpT = wqT + (size_t)C3 * CIN;   // [768][768]

    // pre-pass: f16 conversions (+ weight transposes for row-copy tile fills)
    cvt_copy     <<<(int)(SZ / 4 / 256), 256, 0, stream>>>(x, xh);
    cvt_transpose<<<(int)((size_t)CIN * C3 / 256), 256, 0, stream>>>(wqkv, wqT, CIN, C3);
    cvt_transpose<<<(int)((size_t)CIN * CIN / 256), 256, 0, stream>>>(wproj, wpT, CIN, CIN);

    qkv_gemm <<<dim3(64, 18), 256, 0, stream>>>(xh, wqT, Qb, Kb, Vt);
    attn_fwd <<<dim3(96, 8),  256, 0, stream>>>(Qb, Kb, Vt, Ab);
    proj_gemm<<<dim3(64, 6),  256, 0, stream>>>(Ab, wpT, bproj, out);
}
